// MoE_2113123910117
// MI455X (gfx1250) — matrix operand order repaired and checked by reading, not compiled, not run on hardware
//
#include <hip/hip_runtime.h>

// ---------------------------------------------------------------------------
// MoE (top-2 of 8 experts, SwiGLU) for MI455X / gfx1250.
// bf16 WMMA (v_wmma_f32_16x16x32_bf16) with fp32 accumulate.
// All GEMM tiles staged with async global->LDS (ASYNCcnt), double-buffered.
// Weights pre-converted fp32->bf16 once per call (streaming kernel).
// ---------------------------------------------------------------------------

typedef __bf16 bf16;
typedef bf16  v16bf __attribute__((ext_vector_type(16)));
typedef float v8f   __attribute__((ext_vector_type(8)));

static constexpr int D_        = 1024;
static constexpr int FF_       = 2048;
static constexpr int E_        = 8;
static constexpr int NTOK      = 4 * 2048;        // 8192 tokens
static constexpr int SLOTS     = 2 * NTOK;        // top-2 -> 16384 routed slots
static constexpr int BM        = 128;             // GEMM row tile (slots)
static constexpr int BK        = 64;              // GEMM k tile
static constexpr int BN        = 64;              // GEMM col tile
static constexpr int MAX_SLOTS = SLOTS + E_ * BM; // 17408 (per-expert pad to BM)
static constexpr int ROW_TILES = MAX_SLOTS / BM;  // 136
static constexpr int LP        = BK + 8;          // LDS pitch in bf16 (144B, 16B-aligned)
static constexpr long WELEMS   = (long)E_ * FF_ * D_;  // elems per weight tensor

union FragBF { v16bf v; uint4 u[2]; };

__device__ __forceinline__ unsigned short f2bf(float f) {
    unsigned int u = __float_as_uint(f);
    unsigned int r = (u + 0x7FFFu + ((u >> 16) & 1u)) >> 16;  // RNE
    return (unsigned short)r;
}
__device__ __forceinline__ unsigned int pack2(float lo, float hi) {
    return (unsigned int)f2bf(lo) | ((unsigned int)f2bf(hi) << 16);
}
__device__ __forceinline__ v8f wmma_bf16(const FragBF& a, const FragBF& b, v8f c) {
    return __builtin_amdgcn_wmma_f32_16x16x32_bf16(false, a.v, false, b.v,
                                                   (short)0, c, false, false);
}

// async global->LDS copies; one IOFFSET advances both LDS and global address
__device__ __forceinline__ void async_tile64(unsigned ldsoff, const void* g) {
    unsigned long long ga = (unsigned long long)g;
    asm volatile(
        "global_load_async_to_lds_b128 %0, %1, off\n\t"
        "global_load_async_to_lds_b128 %0, %1, off offset:16\n\t"
        "global_load_async_to_lds_b128 %0, %1, off offset:32\n\t"
        "global_load_async_to_lds_b128 %0, %1, off offset:48"
        :: "v"(ldsoff), "v"(ga)
        : "memory");
}
__device__ __forceinline__ void async_tile32(unsigned ldsoff, const void* g) {
    unsigned long long ga = (unsigned long long)g;
    asm volatile(
        "global_load_async_to_lds_b128 %0, %1, off\n\t"
        "global_load_async_to_lds_b128 %0, %1, off offset:16"
        :: "v"(ldsoff), "v"(ga)
        : "memory");
}
__device__ __forceinline__ void wait_async() {
    asm volatile("s_wait_asynccnt 0x0" ::: "memory");
}
__device__ __forceinline__ unsigned lds_off(const void* p) {
    return (unsigned)(uintptr_t)p;  // LDS aperture keeps offset in addr[31:0]
}

// ---------------------------------------------------------------------------
// 0) convert: fp32 weights -> bf16 (streaming, once per call)
// ---------------------------------------------------------------------------
__global__ __launch_bounds__(256) void moe_convert(const float* __restrict__ src,
                                                   unsigned short* __restrict__ dst) {
    int i = blockIdx.x * blockDim.x + threadIdx.x;  // over WELEMS/4
    float4 v = ((const float4*)src)[i];
    uint2 o;
    o.x = pack2(v.x, v.y);
    o.y = pack2(v.z, v.w);
    ((uint2*)dst)[i] = o;
}

// ---------------------------------------------------------------------------
// 1) init: zero expert counts, mark all slots unused
// ---------------------------------------------------------------------------
__global__ void moe_init(int* counts, int* slot_token) {
    int i = blockIdx.x * blockDim.x + threadIdx.x;
    if (i < E_) counts[i] = 0;
    if (i < MAX_SLOTS) slot_token[i] = -1;
}

// ---------------------------------------------------------------------------
// 2) router: one wave32 per token. logits -> softmax -> top2 -> norm weights
// ---------------------------------------------------------------------------
__global__ __launch_bounds__(256) void moe_router(const float* __restrict__ x,
                                                  const float* __restrict__ Wr,
                                                  float* __restrict__ tok_w,
                                                  int* __restrict__ tok_idx,
                                                  int* __restrict__ counts) {
    const int t    = blockIdx.x * 8 + (threadIdx.x >> 5);
    const int lane = threadIdx.x & 31;

    float acc[E_];
#pragma unroll
    for (int e = 0; e < E_; ++e) acc[e] = 0.f;

    const float* xr = x + (size_t)t * D_;
    for (int d = lane; d < D_; d += 32) {
        float xv = xr[d];
#pragma unroll
        for (int e = 0; e < E_; ++e) acc[e] += xv * Wr[e * D_ + d];
    }
#pragma unroll
    for (int e = 0; e < E_; ++e) {
#pragma unroll
        for (int off = 16; off > 0; off >>= 1) acc[e] += __shfl_xor(acc[e], off);
    }
    if (lane == 0) {
        float m = acc[0];
#pragma unroll
        for (int e = 1; e < E_; ++e) m = fmaxf(m, acc[e]);
        float p[E_], sum = 0.f;
#pragma unroll
        for (int e = 0; e < E_; ++e) { p[e] = __expf(acc[e] - m); sum += p[e]; }
        float inv = 1.f / sum;
#pragma unroll
        for (int e = 0; e < E_; ++e) p[e] *= inv;
        int i0 = 0;
#pragma unroll
        for (int e = 1; e < E_; ++e) if (p[e] > p[i0]) i0 = e;
        int i1 = (i0 == 0) ? 1 : 0;
#pragma unroll
        for (int e = 0; e < E_; ++e) if (e != i0 && p[e] > p[i1]) i1 = e;
        float denom = p[i0] + p[i1] + 1e-8f;
        tok_idx[2 * t]     = i0;
        tok_idx[2 * t + 1] = i1;
        tok_w[2 * t]       = p[i0] / denom;
        tok_w[2 * t + 1]   = p[i1] / denom;
        atomicAdd(&counts[i0], 1);
        atomicAdd(&counts[i1], 1);
    }
}

// ---------------------------------------------------------------------------
// 3) scan: padded per-expert slot offsets (each range multiple of BM)
// ---------------------------------------------------------------------------
__global__ void moe_scan(const int* counts, int* poff, int* cursors) {
    if (threadIdx.x == 0 && blockIdx.x == 0) {
        int acc = 0;
        for (int e = 0; e < E_; ++e) {
            poff[e]    = acc;
            cursors[e] = acc;
            acc += ((counts[e] + BM - 1) / BM) * BM;
        }
        poff[E_] = acc;
    }
}

// ---------------------------------------------------------------------------
// 4) scatter: token -> slot (and inverse map for the combine pass)
// ---------------------------------------------------------------------------
__global__ void moe_scatter(const int* __restrict__ tok_idx, int* cursors,
                            int* __restrict__ slot_token,
                            int* __restrict__ token_slots) {
    int i = blockIdx.x * blockDim.x + threadIdx.x;
    if (i >= SLOTS) return;
    int e = tok_idx[i];
    int s = atomicAdd(&cursors[e], 1);
    slot_token[s]  = i >> 1;
    token_slots[i] = s;
}

// ---------------------------------------------------------------------------
// 5) gather: Xg[slot] = bf16(x[token]) (zeros for padding slots)
// ---------------------------------------------------------------------------
__global__ __launch_bounds__(256) void moe_gather(const float* __restrict__ x,
                                                  const int* __restrict__ slot_token,
                                                  const int* __restrict__ poff,
                                                  unsigned short* __restrict__ Xg) {
    int s = blockIdx.x;
    if (s >= poff[E_]) return;
    int t = slot_token[s];
    int d = threadIdx.x * 4;
    unsigned short* dst = Xg + (size_t)s * D_ + d;
    if (t < 0) {
        *(uint2*)dst = make_uint2(0u, 0u);
    } else {
        float4 v = *(const float4*)(x + (size_t)t * D_ + d);
        uint2 o;
        o.x = pack2(v.x, v.y);
        o.y = pack2(v.z, v.w);
        *(uint2*)dst = o;
    }
}

// ---------------------------------------------------------------------------
// 6) gate/up GEMM + SwiGLU:  H[s,f] = silu(Xg·Wg[e]^T) * (Xg·Wu[e]^T)
//    block = 128 slots x 64 f-cols, 8 waves (4x2), wave = 32x32, BK=64
//    double-buffered LDS; ALL tiles via async global->LDS (ASYNCcnt)
// ---------------------------------------------------------------------------
__global__ __launch_bounds__(256) void moe_gateup(const unsigned short* __restrict__ Xg,
                                                  const unsigned short* __restrict__ Wg16,
                                                  const unsigned short* __restrict__ Wu16,
                                                  unsigned short* __restrict__ H,
                                                  const int* __restrict__ poff) {
    constexpr int ABUF = BM * LP;  // elems
    constexpr int BBUF = BN * LP;
    __shared__ __align__(16) unsigned short sA[2 * ABUF];
    __shared__ __align__(16) unsigned short sG[2 * BBUF];
    __shared__ __align__(16) unsigned short sU[2 * BBUF];

    const int total = poff[E_];
    const int row0  = blockIdx.y * BM;
    if (row0 >= total) return;
    int e = 0;
#pragma unroll
    for (int i = 0; i < E_; ++i) if (row0 >= poff[i + 1]) e = i + 1;

    const int f0 = blockIdx.x * BN;
    const unsigned short* wg = Wg16 + ((size_t)e * FF_ + f0) * D_;
    const unsigned short* wu = Wu16 + ((size_t)e * FF_ + f0) * D_;

    const int tid  = threadIdx.x;
    const int lane = tid & 31;
    const int wid  = tid >> 5;
    const int wm   = wid & 3;   // 4 row groups of 32
    const int wn   = wid >> 2;  // 2 col groups of 32

    v8f accG[2][2], accU[2][2];
#pragma unroll
    for (int mi = 0; mi < 2; ++mi)
#pragma unroll
        for (int ni = 0; ni < 2; ++ni) {
            accG[mi][ni] = (v8f){0.f, 0.f, 0.f, 0.f, 0.f, 0.f, 0.f, 0.f};
            accU[mi][ni] = (v8f){0.f, 0.f, 0.f, 0.f, 0.f, 0.f, 0.f, 0.f};
        }

    const int ar = tid >> 1;          // A staging: row 0..127, 64B strip
    const int ac = (tid & 1) * 32;    // elem offset 0 / 32
    const int br = tid >> 2;          // B staging: row 0..63, 32B strip
    const int bc = (tid & 3) * 16;    // elem offset 0/16/32/48

    const unsigned short* aRow = Xg + (size_t)(row0 + ar) * D_ + ac;
    const unsigned short* gRow = wg + (size_t)br * D_ + bc;
    const unsigned short* uRow = wu + (size_t)br * D_ + bc;
    const unsigned aLds = lds_off(sA + ar * LP + ac);
    const unsigned gLds = lds_off(sG + br * LP + bc);
    const unsigned uLds = lds_off(sU + br * LP + bc);

    // ---- prologue: stage tile 0 into buffer 0 ----
    async_tile64(aLds, aRow);
    async_tile32(gLds, gRow);
    async_tile32(uLds, uRow);
    wait_async();
    __syncthreads();

    constexpr int KT = D_ / BK;  // 16
    for (int kt = 0; kt < KT; ++kt) {
        const int cur  = kt & 1;
        const int nxt  = cur ^ 1;
        const bool more = (kt + 1) < KT;
        if (more) {
            const int kn = (kt + 1) * BK;
            async_tile64(aLds + (unsigned)(nxt * ABUF * 2), aRow + kn);
            async_tile32(gLds + (unsigned)(nxt * BBUF * 2), gRow + kn);
            async_tile32(uLds + (unsigned)(nxt * BBUF * 2), uRow + kn);
        }
        // ---- compute tile kt ----
        const unsigned short* cA = sA + cur * ABUF;
        const unsigned short* cG = sG + cur * BBUF;
        const unsigned short* cU = sU + cur * BBUF;
#pragma unroll
        for (int ks = 0; ks < BK; ks += 32) {
            FragBF afr[2], gfr[2], ufr[2];
#pragma unroll
            for (int mi = 0; mi < 2; ++mi) {
                int base = (wm * 32 + mi * 16 + (lane & 15)) * LP + ks + ((lane >> 4) * 16);
                afr[mi].u[0] = *(const uint4*)(cA + base);
                afr[mi].u[1] = *(const uint4*)(cA + base + 8);
            }
#pragma unroll
            for (int ni = 0; ni < 2; ++ni) {
                int base = (wn * 32 + ni * 16 + (lane & 15)) * LP + ks + ((lane >> 4) * 16);
                gfr[ni].u[0] = *(const uint4*)(cG + base);
                gfr[ni].u[1] = *(const uint4*)(cG + base + 8);
                ufr[ni].u[0] = *(const uint4*)(cU + base);
                ufr[ni].u[1] = *(const uint4*)(cU + base + 8);
            }
#pragma unroll
            for (int mi = 0; mi < 2; ++mi)
#pragma unroll
                for (int ni = 0; ni < 2; ++ni) {
                    accG[mi][ni] = wmma_bf16(afr[mi], gfr[ni], accG[mi][ni]);
                    accU[mi][ni] = wmma_bf16(afr[mi], ufr[ni], accU[mi][ni]);
                }
        }
        if (more) {
            wait_async();
            __syncthreads();
        }
    }

    // ---- SwiGLU epilogue -> bf16 H ----
    const int lHi = (lane >> 4) * 8;
    const int ln  = lane & 15;
#pragma unroll
    for (int mi = 0; mi < 2; ++mi)
#pragma unroll
        for (int ni = 0; ni < 2; ++ni)
#pragma unroll
            for (int r = 0; r < 8; ++r) {
                float g = accG[mi][ni][r];
                float u = accU[mi][ni][r];
                float h = (g / (1.f + __expf(-g))) * u;
                int row = row0 + wm * 32 + mi * 16 + lHi + r;
                int col = f0 + wn * 32 + ni * 16 + ln;
                H[(size_t)row * FF_ + col] = f2bf(h);
            }
}

// ---------------------------------------------------------------------------
// 7) down GEMM:  Y[s,d] = sum_f H[s,f] * Wd[e][d,f]   (same pipeline)
// ---------------------------------------------------------------------------
__global__ __launch_bounds__(256) void moe_down(const unsigned short* __restrict__ H,
                                                const unsigned short* __restrict__ Wd16,
                                                float* __restrict__ Yg,
                                                const int* __restrict__ poff) {
    constexpr int ABUF = BM * LP;
    constexpr int BBUF = BN * LP;
    __shared__ __align__(16) unsigned short sA[2 * ABUF];
    __shared__ __align__(16) unsigned short sB[2 * BBUF];

    const int total = poff[E_];
    const int row0  = blockIdx.y * BM;
    if (row0 >= total) return;
    int e = 0;
#pragma unroll
    for (int i = 0; i < E_; ++i) if (row0 >= poff[i + 1]) e = i + 1;

    const int d0 = blockIdx.x * BN;
    const unsigned short* wd = Wd16 + ((size_t)e * D_ + d0) * FF_;

    const int tid  = threadIdx.x;
    const int lane = tid & 31;
    const int wid  = tid >> 5;
    const int wm   = wid & 3;
    const int wn   = wid >> 2;

    v8f acc[2][2];
#pragma unroll
    for (int mi = 0; mi < 2; ++mi)
#pragma unroll
        for (int ni = 0; ni < 2; ++ni)
            acc[mi][ni] = (v8f){0.f, 0.f, 0.f, 0.f, 0.f, 0.f, 0.f, 0.f};

    const int ar = tid >> 1;
    const int ac = (tid & 1) * 32;
    const int br = tid >> 2;
    const int bc = (tid & 3) * 16;

    const unsigned short* aRow = H + (size_t)(row0 + ar) * FF_ + ac;
    const unsigned short* bRow = wd + (size_t)br * FF_ + bc;
    const unsigned aLds = lds_off(sA + ar * LP + ac);
    const unsigned bLds = lds_off(sB + br * LP + bc);

    async_tile64(aLds, aRow);
    async_tile32(bLds, bRow);
    wait_async();
    __syncthreads();

    constexpr int KT = FF_ / BK;  // 32
    for (int kt = 0; kt < KT; ++kt) {
        const int cur  = kt & 1;
        const int nxt  = cur ^ 1;
        const bool more = (kt + 1) < KT;
        if (more) {
            const int kn = (kt + 1) * BK;
            async_tile64(aLds + (unsigned)(nxt * ABUF * 2), aRow + kn);
            async_tile32(bLds + (unsigned)(nxt * BBUF * 2), bRow + kn);
        }
        const unsigned short* cA = sA + cur * ABUF;
        const unsigned short* cB = sB + cur * BBUF;
#pragma unroll
        for (int ks = 0; ks < BK; ks += 32) {
            FragBF afr[2], bfr[2];
#pragma unroll
            for (int mi = 0; mi < 2; ++mi) {
                int base = (wm * 32 + mi * 16 + (lane & 15)) * LP + ks + ((lane >> 4) * 16);
                afr[mi].u[0] = *(const uint4*)(cA + base);
                afr[mi].u[1] = *(const uint4*)(cA + base + 8);
            }
#pragma unroll
            for (int ni = 0; ni < 2; ++ni) {
                int base = (wn * 32 + ni * 16 + (lane & 15)) * LP + ks + ((lane >> 4) * 16);
                bfr[ni].u[0] = *(const uint4*)(cB + base);
                bfr[ni].u[1] = *(const uint4*)(cB + base + 8);
            }
#pragma unroll
            for (int mi = 0; mi < 2; ++mi)
#pragma unroll
                for (int ni = 0; ni < 2; ++ni)
                    acc[mi][ni] = wmma_bf16(afr[mi], bfr[ni], acc[mi][ni]);
        }
        if (more) {
            wait_async();
            __syncthreads();
        }
    }

    const int lHi = (lane >> 4) * 8;
    const int ln  = lane & 15;
#pragma unroll
    for (int mi = 0; mi < 2; ++mi)
#pragma unroll
        for (int ni = 0; ni < 2; ++ni)
#pragma unroll
            for (int r = 0; r < 8; ++r) {
                int row = row0 + wm * 32 + mi * 16 + lHi + r;
                int col = d0 + wn * 32 + ni * 16 + ln;
                Yg[(size_t)row * D_ + col] = acc[mi][ni][r];
            }
}

// ---------------------------------------------------------------------------
// 8) combine: out[t] = w0*Y[slot0] + w1*Y[slot1]   (full overwrite, no atomics)
// ---------------------------------------------------------------------------
__global__ __launch_bounds__(256) void moe_combine(const float* __restrict__ Yg,
                                                   const int* __restrict__ token_slots,
                                                   const float* __restrict__ tok_w,
                                                   float* __restrict__ out) {
    int idx = blockIdx.x * blockDim.x + threadIdx.x;  // NTOK * (D/4)
    int t = idx >> 8;       // D/4 = 256
    int j = idx & 255;
    int s0 = token_slots[2 * t];
    int s1 = token_slots[2 * t + 1];
    float w0 = tok_w[2 * t];
    float w1 = tok_w[2 * t + 1];
    float4 a = ((const float4*)(Yg + (size_t)s0 * D_))[j];
    float4 b = ((const float4*)(Yg + (size_t)s1 * D_))[j];
    float4 o;
    o.x = w0 * a.x + w1 * b.x;
    o.y = w0 * a.y + w1 * b.y;
    o.z = w0 * a.z + w1 * b.z;
    o.w = w0 * a.w + w1 * b.w;
    ((float4*)(out + (size_t)t * D_))[j] = o;
}

// ---------------------------------------------------------------------------
extern "C" void kernel_launch(void* const* d_in, const int* in_sizes, int n_in,
                              void* d_out, int out_size, void* d_ws, size_t ws_size,
                              hipStream_t stream) {
    (void)in_sizes; (void)n_in; (void)out_size; (void)ws_size;
    const float* x  = (const float*)d_in[0];
    const float* Wr = (const float*)d_in[1];
    const float* Wg = (const float*)d_in[2];
    const float* Wu = (const float*)d_in[3];
    const float* Wd = (const float*)d_in[4];
    float* out = (float*)d_out;

    char* w = (char*)d_ws;
    size_t off = 0;
    auto take = [&](size_t bytes) {
        size_t cur = off;
        off += (bytes + 255) & ~(size_t)255;
        return cur;
    };
    int*   counts      = (int*)(w + take(E_ * 4));
    int*   cursors     = (int*)(w + take(E_ * 4));
    int*   poff        = (int*)(w + take((E_ + 1) * 4));
    int*   tok_idx     = (int*)(w + take((size_t)SLOTS * 4));
    float* tok_w       = (float*)(w + take((size_t)SLOTS * 4));
    int*   token_slots = (int*)(w + take((size_t)SLOTS * 4));
    int*   slot_token  = (int*)(w + take((size_t)MAX_SLOTS * 4));
    unsigned short* Xg   = (unsigned short*)(w + take((size_t)MAX_SLOTS * D_ * 2));
    unsigned short* H    = (unsigned short*)(w + take((size_t)MAX_SLOTS * FF_ * 2));
    float*          Yg   = (float*)(w + take((size_t)MAX_SLOTS * D_ * 4));
    unsigned short* Wg16 = (unsigned short*)(w + take((size_t)WELEMS * 2));
    unsigned short* Wu16 = (unsigned short*)(w + take((size_t)WELEMS * 2));
    unsigned short* Wd16 = (unsigned short*)(w + take((size_t)WELEMS * 2));

    const int cvtBlocks = (int)(WELEMS / 4 / 256);  // 16384
    moe_convert<<<cvtBlocks, 256, 0, stream>>>(Wg, Wg16);
    moe_convert<<<cvtBlocks, 256, 0, stream>>>(Wu, Wu16);
    moe_convert<<<cvtBlocks, 256, 0, stream>>>(Wd, Wd16);

    moe_init<<<(MAX_SLOTS + 255) / 256, 256, 0, stream>>>(counts, slot_token);
    moe_router<<<NTOK / 8, 256, 0, stream>>>(x, Wr, tok_w, tok_idx, counts);
    moe_scan<<<1, 32, 0, stream>>>(counts, poff, cursors);
    moe_scatter<<<(SLOTS + 255) / 256, 256, 0, stream>>>(tok_idx, cursors, slot_token, token_slots);
    moe_gather<<<MAX_SLOTS, 256, 0, stream>>>(x, slot_token, poff, Xg);
    moe_gateup<<<dim3(FF_ / BN, ROW_TILES), 256, 0, stream>>>(Xg, Wg16, Wu16, H, poff);
    moe_down<<<dim3(D_ / BN, ROW_TILES), 256, 0, stream>>>(H, Wd16, Yg, poff);
    moe_combine<<<(NTOK * (D_ / 4)) / 256, 256, 0, stream>>>(Yg, token_slots, tok_w, out);
}
